// ddstgcn_13125420056716
// MI455X (gfx1250) — compile-verified, matrix operand order checked by reading, not run on hardware
//
#include <hip/hip_runtime.h>

// ---------------- dimensions ----------------
#define B_   4
#define N_   512
#define T0   13
#define E_   1024
#define C_   40
#define HID_ 20
#define SK_  320
#define EC_  640
#define OD_  12
#define LYR_ 3

typedef _Float16 v16h __attribute__((ext_vector_type(16)));
typedef float    v8f  __attribute__((ext_vector_type(8)));

// =====================================================================
// Generic WMMA GEMM:  C[m,n] (+)= act( sum_k A(m,k)*scale(k)*B(k,n) + bias[m] )
// A index = mo*sAo + mi*sAi + k*sAk         (m = mo*mInner + mi)
// B index = no*sBo + ni*sBi + k*sBk         (n = no*nInner + ni)
// C index = mo*sCmo + mi*sCmi + no*sCno + ni*sCni
//
// One wave -> one 16x32 tile (two WMMAs sharing the A fragment);
// 4 waves/block -> 16x128 block tile.
// Bounds: m/n are CLAMPED (garbage confined to unstored rows/cols, no
// branches); K tail masked with cndmask only when KM (K%32!=0).
// AV/BV: unit-k-stride vector path, 4x global_load_b128 per fragment.
// =====================================================================
template <bool AV, bool BV, bool KM, bool SCALE>
__global__ __launch_bounds__(128) void k_gemm_wmma(
    const float* __restrict__ A, const float* __restrict__ Bm,
    float* __restrict__ Cm, const float* __restrict__ bias,
    const float* __restrict__ kscale,
    int M, int Ncols, int K,
    long aB, long bB, long cB,
    int mInner, long sAo, long sAi, long sAk,
    int nInner, long sBo, long sBi, long sBk,
    long sCmo, long sCmi, long sCno, long sCni,
    int act, int accum)
{
    const int lane = threadIdx.x & 31;
    const int wave = threadIdx.x >> 5;
    const int tm0  = blockIdx.x * 16;
    const int tn0  = (blockIdx.y * 4 + wave) * 32;
    const long batch = blockIdx.z;

    const float* Ab = A  + batch * aB;
    const float* Bb = Bm + batch * bB;
    float*       Cb = Cm + batch * cB;

    const int lm   = lane & 15;
    const int half = lane >> 4;

    // clamped fragment row / cols (no branches, garbage never stored)
    int am  = tm0 + lm;        if (am  > M - 1)     am  = M - 1;
    int bn0 = tn0 + lm;        if (bn0 > Ncols - 1) bn0 = Ncols - 1;
    int bn1 = tn0 + 16 + lm;   if (bn1 > Ncols - 1) bn1 = Ncols - 1;

    long aRow, bCol0, bCol1;
    { int mo = am  / mInner, mi = am  % mInner; aRow  = (long)mo * sAo + (long)mi * sAi; }
    { int no = bn0 / nInner, ni = bn0 % nInner; bCol0 = (long)no * sBo + (long)ni * sBi; }
    { int no = bn1 / nInner, ni = bn1 % nInner; bCol1 = (long)no * sBo + (long)ni * sBi; }

    v8f acc0 = {}, acc1 = {};

    for (int kb = 0; kb < K; kb += 32) {
        float af[16], bf0[16], bf1[16], sc[16];

        if (AV) {
            const float* base = Ab + aRow + kb + 8 * half;
            float4 a0 = *(const float4*)(base);
            float4 a1 = *(const float4*)(base + 4);
            float4 a2 = *(const float4*)(base + 16);
            float4 a3 = *(const float4*)(base + 20);
            af[0]=a0.x; af[1]=a0.y; af[2]=a0.z;  af[3]=a0.w;
            af[4]=a1.x; af[5]=a1.y; af[6]=a1.z;  af[7]=a1.w;
            af[8]=a2.x; af[9]=a2.y; af[10]=a2.z; af[11]=a2.w;
            af[12]=a3.x; af[13]=a3.y; af[14]=a3.z; af[15]=a3.w;
        } else {
#pragma unroll
            for (int j = 0; j < 16; ++j) {
                int k = kb + j + 8 * ((j >> 3) + half);
                int kc = KM ? (k > K - 1 ? K - 1 : k) : k;
                af[j] = Ab[aRow + (long)kc * sAk];
            }
        }

        if (BV) {
            const float* b0p = Bb + bCol0 + kb + 8 * half;
            const float* b1p = Bb + bCol1 + kb + 8 * half;
            float4 u0 = *(const float4*)(b0p);
            float4 u1 = *(const float4*)(b0p + 4);
            float4 u2 = *(const float4*)(b0p + 16);
            float4 u3 = *(const float4*)(b0p + 20);
            bf0[0]=u0.x; bf0[1]=u0.y; bf0[2]=u0.z;  bf0[3]=u0.w;
            bf0[4]=u1.x; bf0[5]=u1.y; bf0[6]=u1.z;  bf0[7]=u1.w;
            bf0[8]=u2.x; bf0[9]=u2.y; bf0[10]=u2.z; bf0[11]=u2.w;
            bf0[12]=u3.x; bf0[13]=u3.y; bf0[14]=u3.z; bf0[15]=u3.w;
            float4 w0 = *(const float4*)(b1p);
            float4 w1 = *(const float4*)(b1p + 4);
            float4 w2 = *(const float4*)(b1p + 16);
            float4 w3 = *(const float4*)(b1p + 20);
            bf1[0]=w0.x; bf1[1]=w0.y; bf1[2]=w0.z;  bf1[3]=w0.w;
            bf1[4]=w1.x; bf1[5]=w1.y; bf1[6]=w1.z;  bf1[7]=w1.w;
            bf1[8]=w2.x; bf1[9]=w2.y; bf1[10]=w2.z; bf1[11]=w2.w;
            bf1[12]=w3.x; bf1[13]=w3.y; bf1[14]=w3.z; bf1[15]=w3.w;
        } else {
#pragma unroll
            for (int j = 0; j < 16; ++j) {
                int k = kb + j + 8 * ((j >> 3) + half);
                int kc = KM ? (k > K - 1 ? K - 1 : k) : k;
                long ko = (long)kc * sBk;
                bf0[j] = Bb[bCol0 + ko];
                bf1[j] = Bb[bCol1 + ko];
            }
        }

        if (SCALE) {
#pragma unroll
            for (int j = 0; j < 16; ++j) {
                int k = kb + j + 8 * ((j >> 3) + half);
                int kc = KM ? (k > K - 1 ? K - 1 : k) : k;
                sc[j] = kscale[kc];
            }
        }

        v16h av, bv0, bv1;
#pragma unroll
        for (int j = 0; j < 16; ++j) {
            float a = af[j], b0 = bf0[j], b1 = bf1[j];
            if (KM) {
                int k = kb + j + 8 * ((j >> 3) + half);
                bool kin = (k < K);
                a  = kin ? a  : 0.0f;
                b0 = kin ? b0 : 0.0f;
                b1 = kin ? b1 : 0.0f;
            }
            if (SCALE) { b0 *= sc[j]; b1 *= sc[j]; }
            av[j]  = (_Float16)a;
            bv0[j] = (_Float16)b0;
            bv1[j] = (_Float16)b1;
        }
        acc0 = __builtin_amdgcn_wmma_f32_16x16x32_f16(
            false, av, false, bv0, (short)0, acc0, false, false);
        acc1 = __builtin_amdgcn_wmma_f32_16x16x32_f16(
            false, av, false, bv1, (short)0, acc1, false, false);
    }

#pragma unroll
    for (int r = 0; r < 8; ++r) {
        int m = tm0 + r + 8 * half;
        if (m >= M) continue;
        int mo = m / mInner, mi = m % mInner;
        long cm = (long)mo * sCmo + (long)mi * sCmi;
        float vb = bias ? bias[m] : 0.0f;
#pragma unroll
        for (int tile = 0; tile < 2; ++tile) {
            int n = tn0 + tile * 16 + lm;
            if (n >= Ncols) continue;
            int no = n / nInner, ni = n % nInner;
            long ci = cm + (long)no * sCno + (long)ni * sCni;
            float v = (tile == 0 ? acc0[r] : acc1[r]) + vb;
            if (accum) v += Cb[ci];
            if (act == 1) v = fmaxf(v, 0.0f);
            Cb[ci] = v;
        }
    }
}

static inline void gemm(hipStream_t st,
    const float* A, const float* Bm, float* Cm,
    const float* bias, const float* kscale,
    int M, int Ncols, int K, int batches,
    long aB, long bB, long cB,
    int mInner, long sAo, long sAi, long sAk,
    int nInner, long sBo, long sBi, long sBk,
    long sCmo, long sCmi, long sCno, long sCni,
    int act, int accum)
{
    dim3 g((M + 15) / 16, (Ncols + 127) / 128, batches);
    bool kfull = (K % 32) == 0;
    bool av = (sAk == 1) && kfull;
    bool bv = (sBk == 1) && kfull;
    bool km = !kfull;
    bool sc = (kscale != nullptr);
#define GEMM_ARGS A, Bm, Cm, bias, kscale, M, Ncols, K, aB, bB, cB, \
    mInner, sAo, sAi, sAk, nInner, sBo, sBi, sBk, sCmo, sCmi, sCno, sCni, act, accum
#define GO(AVv, BVv, KMv, SCv) \
    k_gemm_wmma<AVv, BVv, KMv, SCv><<<g, dim3(128, 1, 1), 0, st>>>(GEMM_ARGS)
    if (av) {
        if (sc) { if (km) GO(true,false,true,true);  else GO(true,false,false,true); }
        else    { if (km) GO(true,false,true,false); else GO(true,false,false,false); }
    } else if (bv) {
        if (sc) { if (km) GO(false,true,true,true);  else GO(false,true,false,true); }
        else    { if (km) GO(false,true,true,false); else GO(false,true,false,false); }
    } else {
        if (sc) { if (km) GO(false,false,true,true);  else GO(false,false,false,true); }
        else    { if (km) GO(false,false,true,false); else GO(false,false,false,false); }
    }
#undef GO
#undef GEMM_ARGS
}

// ---------------- small elementwise / reduction kernels ----------------

__global__ void k_start(const float* __restrict__ x, const float* __restrict__ sw,
                        const float* __restrict__ sb, float* __restrict__ h, int tot)
{
    int i = blockIdx.x * blockDim.x + threadIdx.x;
    if (i >= tot) return;
    int t = i % T0;
    int n = (i / T0) % N_;
    int c = (i / (T0 * N_)) % C_;
    int b = i / (T0 * N_ * C_);
    h[i] = sw[c] * x[(long)b * N_ * T0 + (long)n * T0 + t] + sb[c];
}

// adp = softmax(relu(nv1 @ nv2), axis=1)  -- one block per row
__global__ void k_adj(const float* __restrict__ nv1, const float* __restrict__ nv2,
                      float* __restrict__ adp)
{
    __shared__ float red[256];
    int r = blockIdx.x, tc = threadIdx.x;
    float v[2];
#pragma unroll
    for (int s = 0; s < 2; ++s) {
        int c = tc + s * 256;
        float acc = 0.f;
#pragma unroll
        for (int j = 0; j < 10; ++j) acc += nv1[r * 10 + j] * nv2[j * N_ + c];
        v[s] = fmaxf(acc, 0.f);
    }
    red[tc] = fmaxf(v[0], v[1]); __syncthreads();
    for (int off = 128; off > 0; off >>= 1) {
        if (tc < off) red[tc] = fmaxf(red[tc], red[tc + off]);
        __syncthreads();
    }
    float mx = red[0]; __syncthreads();
    float e0 = expf(v[0] - mx), e1 = expf(v[1] - mx);
    red[tc] = e0 + e1; __syncthreads();
    for (int off = 128; off > 0; off >>= 1) {
        if (tc < off) red[tc] += red[tc + off];
        __syncthreads();
    }
    float inv = 1.0f / red[0];
    adp[(long)r * N_ + tc]       = e0 * inv;
    adp[(long)r * N_ + tc + 256] = e1 * inv;
}

__global__ void k_zero(float* __restrict__ p, int n)
{
    int i = blockIdx.x * blockDim.x + threadIdx.x;
    if (i < n) p[i] = 0.f;
}

__global__ void k_relu(float* __restrict__ p, int n)
{
    int i = blockIdx.x * blockDim.x + threadIdx.x;
    if (i < n) p[i] = fmaxf(p[i], 0.f);
}

// lhsv[b,n,c] = sum_t h[b,c,n,t]*(t/(T-1))*w1 ; rhsv uses (1 - t/(T-1))*w2
__global__ void k_timeproj(const float* __restrict__ h, float* __restrict__ lhsv,
                           float* __restrict__ rhsv, const float* __restrict__ w1p,
                           const float* __restrict__ w2p, int T)
{
    int bn = blockIdx.x;          // b*N + n
    int c = threadIdx.x;
    if (c >= C_) return;
    int b = bn / N_, n = bn % N_;
    float w1 = *w1p, w2 = *w2p;
    const float* hp = h + (((long)b * C_ + c) * N_ + n) * T;
    float s1 = 0.f, s2 = 0.f, inv = 1.0f / (float)(T - 1);
    for (int t = 0; t < T; ++t) {
        float f = (float)t * inv;
        float hv = hp[t];
        s1 += hv * f * w1;
        s2 += hv * (1.0f - f) * w2;
    }
    lhsv[(long)bn * C_ + c] = s1;
    rhsv[(long)bn * C_ + c] = s2;
}

// ef[b,e,c] = ob[c] + sum_j lhsP[b,i0[e],j]*ow[c,j] + sum_j rhsP[b,i1[e],j]*ow[c,H+j]
__global__ void k_satfeat(const float* __restrict__ lhsP, const float* __restrict__ rhsP,
                          const int* __restrict__ i0, const int* __restrict__ i1,
                          const float* __restrict__ ow, const float* __restrict__ ob,
                          float* __restrict__ ef)
{
    int i = blockIdx.x * blockDim.x + threadIdx.x;
    if (i >= B_ * E_ * C_) return;
    int c = i % C_;
    int e = (i / C_) % E_;
    int b = i / (C_ * E_);
    const float* lp = lhsP + ((long)b * N_ + i0[e]) * HID_;
    const float* rp = rhsP + ((long)b * N_ + i1[e]) * HID_;
    const float* wc = ow + (long)c * 2 * HID_;
    float s = ob[c];
#pragma unroll
    for (int j = 0; j < HID_; ++j) s += lp[j] * wc[j] + rp[j] * wc[HID_ + j];
    ef[i] = s;
}

// v[b,e] = sigmoid( sum_c hw[c]*ef[b,e,c] + sum_c hw[C+c]*xw[b,c,e] + hb )
__global__ void k_hgout(const float* __restrict__ ef, const float* __restrict__ xw,
                        const float* __restrict__ hw, const float* __restrict__ hbp,
                        float* __restrict__ v)
{
    int i = blockIdx.x * blockDim.x + threadIdx.x;
    if (i >= B_ * E_) return;
    int e = i % E_, b = i / E_;
    float s = *hbp;
    const float* efp = ef + (long)i * C_;
    const float* xwp = xw + (long)b * C_ * E_ + e;
#pragma unroll 8
    for (int c = 0; c < C_; ++c) s += hw[c] * efp[c] + hw[C_ + c] * xwp[(long)c * E_];
    v[i] = 1.0f / (1.0f + expf(-s));
}

// adpF/adpB [b,v,w] = (v==w) ? adp[v,v] : 0
__global__ void k_diagfill(const float* __restrict__ adp, float* __restrict__ aF,
                           float* __restrict__ aB)
{
    int i = blockIdx.x * blockDim.x + threadIdx.x;
    if (i >= B_ * N_ * N_) return;
    int w = i % N_, v = (i / N_) % N_;
    float val = (v == w) ? adp[(long)v * N_ + v] : 0.f;
    aF[i] = val; aB[i] = val;
}

// scatter edge gates: adpF[b,ix,iy] = adp[ix,iy]*vf ; adpB[b,iy,ix] = adp[ix,iy]*vb
__global__ void k_scatter(float* __restrict__ aF, float* __restrict__ aB,
                          const float* __restrict__ adp, const int* __restrict__ idx,
                          const int* __restrict__ idy, const float* __restrict__ vf,
                          const float* __restrict__ vb)
{
    int i = blockIdx.x * blockDim.x + threadIdx.x;
    if (i >= B_ * E_) return;
    int e = i % E_, b = i / E_;
    int ix = idx[e], iy = idy[e];
    float a = adp[(long)ix * N_ + iy];
    aF[(long)b * N_ * N_ + (long)ix * N_ + iy] = a * vf[i];
    aB[(long)b * N_ * N_ + (long)iy * N_ + ix] = a * vb[i];
}

// fused tanh(filt)*sigmoid(gate) dilated (1,2) conv, dilation 2 -> zcat group 0
__global__ void k_filtgate(const float* __restrict__ h,
                           const float* __restrict__ fw, const float* __restrict__ fb,
                           const float* __restrict__ gw, const float* __restrict__ gb,
                           float* __restrict__ zcat, int T, int Tp)
{
    int i = blockIdx.x * blockDim.x + threadIdx.x;
    if (i >= B_ * C_ * N_ * Tp) return;
    int t = i % Tp;
    int n = (i / Tp) % N_;
    int c = (i / (Tp * N_)) % C_;
    int b = i / (Tp * N_ * C_);
    float f = fb[c], g = gb[c];
    for (int cc = 0; cc < C_; ++cc) {
        const float* hp = h + (((long)b * C_ + cc) * N_ + n) * T + t;
        float h0 = hp[0], h2 = hp[2];
        int wi = (c * C_ + cc) * 2;
        f += fw[wi] * h0 + fw[wi + 1] * h2;
        g += gw[wi] * h0 + gw[wi + 1] * h2;
    }
    float z = tanhf(f) * (1.0f / (1.0f + expf(-g)));
    zcat[(((long)b * 7 * C_ + c) * N_ + n) * Tp + t] = z;
}

// dg conv: (1,2) dilated conv over 7C channels, zcat -> zdg
__global__ void k_dg(const float* __restrict__ zcat, const float* __restrict__ dgw,
                     const float* __restrict__ dgb, float* __restrict__ zdg,
                     int Tp, int Tpp)
{
    int i = blockIdx.x * blockDim.x + threadIdx.x;
    if (i >= B_ * C_ * N_ * Tpp) return;
    int t = i % Tpp;
    int n = (i / Tpp) % N_;
    int c = (i / (Tpp * N_)) % C_;
    int b = i / (Tpp * N_ * C_);
    float s = dgb[c];
    const float* wrow = dgw + (long)c * 7 * C_ * 2;
    const float* zb = zcat + ((long)b * 7 * C_) * N_ * Tp + (long)n * Tp + t;
    for (int q = 0; q < 7 * C_; ++q) {
        const float* zp = zb + (long)q * N_ * Tp;
        s += wrow[q * 2] * zp[0] + wrow[q * 2 + 1] * zp[2];
    }
    zdg[i] = s;
}

// h_next[b,c,n,t] = zdg[b,c,n,t] + h_cur[b,c,n,t+4]
__global__ void k_residual(const float* __restrict__ zdg, const float* __restrict__ hcur,
                           float* __restrict__ hnext, int T, int Tpp)
{
    int i = blockIdx.x * blockDim.x + threadIdx.x;
    if (i >= B_ * C_ * N_ * Tpp) return;
    int t = i % Tpp;
    int n = (i / Tpp) % N_;
    int c = (i / (Tpp * N_)) % C_;
    int b = i / (Tpp * N_ * C_);
    hnext[i] = zdg[i] + hcur[(((long)b * C_ + c) * N_ + n) * T + (t + 4)];
}

// =====================================================================
extern "C" void kernel_launch(void* const* d_in, const int* in_sizes, int n_in,
                              void* d_out, int out_size, void* d_ws, size_t ws_size,
                              hipStream_t stream)
{
    (void)in_sizes; (void)n_in; (void)out_size; (void)ws_size;

    const float* x      = (const float*)d_in[0];
    const int*   idx    = (const int*)d_in[1];
    const int*   idy    = (const int*)d_in[2];
    const float* nv1    = (const float*)d_in[3];
    const float* nv2    = (const float*)d_in[4];
    const float* w_edge[2] = {(const float*)d_in[5], (const float*)d_in[6]};
    const float* G0     = (const float*)d_in[7];
    const float* G1     = (const float*)d_in[8];
    const float* startw = (const float*)d_in[9];
    const float* startb = (const float*)d_in[10];
    const float* sat_w1[2] = {(const float*)d_in[11], (const float*)d_in[19]};
    const float* sat_w2[2] = {(const float*)d_in[12], (const float*)d_in[20]};
    const float* sat_W3[2] = {(const float*)d_in[13], (const float*)d_in[21]};
    const float* sat_W4[2] = {(const float*)d_in[14], (const float*)d_in[22]};
    const float* sat_ow[2] = {(const float*)d_in[15], (const float*)d_in[23]};
    const float* sat_ob[2] = {(const float*)d_in[16], (const float*)d_in[24]};
    const float* hg_w[2]   = {(const float*)d_in[17], (const float*)d_in[25]};
    const float* hg_b[2]   = {(const float*)d_in[18], (const float*)d_in[26]};
    const float* filtw  = (const float*)d_in[27];
    const float* filtb  = (const float*)d_in[28];
    const float* gatew  = (const float*)d_in[29];
    const float* gateb  = (const float*)d_in[30];
    const float* skipw  = (const float*)d_in[31];
    const float* skipb  = (const float*)d_in[32];
    const float* dgw    = (const float*)d_in[33];
    const float* dgb    = (const float*)d_in[34];
    const float* end1w  = (const float*)d_in[35];
    const float* end1b  = (const float*)d_in[36];
    const float* end2w  = (const float*)d_in[37];
    const float* end2b  = (const float*)d_in[38];
    float* out = (float*)d_out;

    // ---- workspace partition (~65 MB of f32) ----
    float* w = (float*)d_ws;
    float* Gdir[2];
    Gdir[0] = w; w += (size_t)E_ * E_;
    Gdir[1] = w; w += (size_t)E_ * E_;
    float* adp  = w; w += (size_t)N_ * N_;
    float* adpF = w; w += (size_t)B_ * N_ * N_;
    float* adpB = w; w += (size_t)B_ * N_ * N_;
    float* h0   = w; w += (size_t)B_ * C_ * N_ * T0;
    float* h1   = w; w += (size_t)B_ * C_ * N_ * T0;
    float* lhsv = w; w += (size_t)B_ * N_ * C_;
    float* rhsv = w; w += (size_t)B_ * N_ * C_;
    float* lhsP = w; w += (size_t)B_ * N_ * HID_;
    float* rhsP = w; w += (size_t)B_ * N_ * HID_;
    float* ef   = w; w += (size_t)B_ * E_ * C_;
    float* xw   = w; w += (size_t)B_ * C_ * E_;
    float* vmat[2];
    vmat[0] = w; w += (size_t)B_ * E_;
    vmat[1] = w; w += (size_t)B_ * E_;
    float* zcat = w; w += (size_t)B_ * 7 * C_ * N_ * 11;
    float* zdg  = w; w += (size_t)B_ * C_ * N_ * 9;
    float* skipf = w; w += (size_t)B_ * SK_ * N_;
    float* e1   = w; w += (size_t)B_ * EC_ * N_;

    // ---- start conv (1x1, CIN=1) ----
    {
        int tot = B_ * C_ * N_ * T0;
        k_start<<<(tot + 255) / 256, 256, 0, stream>>>(x, startw, startb, h0, tot);
    }
    // ---- adjacency softmax ----
    k_adj<<<N_, 256, 0, stream>>>(nv1, nv2, adp);
    // ---- Gf/Gb = G0 @ diag(w_edge) @ G1  (1024^3 WMMA GEMM, scale folded in) ----
    for (int d = 0; d < 2; ++d)
        gemm(stream, G0, G1, Gdir[d], nullptr, w_edge[d],
             E_, E_, E_, 1, 0, 0, 0,
             E_, 0, E_, 1,            // A row-major (unit k-stride -> vector path)
             E_, 0, 1, E_,            // B row-major
             0, E_, 0, 1, 0, 0);
    // ---- zero skip accumulator ----
    k_zero<<<(B_ * SK_ * N_ + 255) / 256, 256, 0, stream>>>(skipf, B_ * SK_ * N_);

    float* hcur = h0;
    float* hnext = h1;
    const int Ts[LYR_] = {13, 9, 5};

    for (int L = 0; L < LYR_; ++L) {
        int T = Ts[L], Tp = T - 2, Tpp = T - 4;

        // ---- spatial attention + hypergraph edge gates (both directions) ----
        for (int d = 0; d < 2; ++d) {
            const int* i0 = (d == 0) ? idx : idy;
            const int* i1 = (d == 0) ? idy : idx;
            k_timeproj<<<B_ * N_, 64, 0, stream>>>(hcur, lhsv, rhsv,
                                                   sat_w1[d] + L, sat_w2[d] + L, T);
            gemm(stream, lhsv, sat_W3[d] + (size_t)L * C_ * HID_, lhsP, nullptr, nullptr,
                 B_ * N_, HID_, C_, 1, 0, 0, 0,
                 B_ * N_, 0, C_, 1,
                 HID_, 0, 1, HID_,
                 0, HID_, 0, 1, 0, 0);
            gemm(stream, rhsv, sat_W4[d] + (size_t)L * C_ * HID_, rhsP, nullptr, nullptr,
                 B_ * N_, HID_, C_, 1, 0, 0, 0,
                 B_ * N_, 0, C_, 1,
                 HID_, 0, 1, HID_,
                 0, HID_, 0, 1, 0, 0);
            {
                int tot = B_ * E_ * C_;
                k_satfeat<<<(tot + 255) / 256, 256, 0, stream>>>(lhsP, rhsP, i0, i1,
                    sat_ow[d] + (size_t)L * C_ * 2 * HID_, sat_ob[d] + (size_t)L * C_, ef);
            }
            // xw[b,c,w] = sum_e ef[b,e,c] * G[w,e]   (M=160, N=K=1024, unit k-stride on B)
            gemm(stream, ef, Gdir[d], xw, nullptr, nullptr,
                 B_ * C_, E_, E_, 1, 0, 0, 0,
                 C_, (long)E_ * C_, 1, C_,     // A: ef[b,e,c] with m=(b,c), k=e
                 E_, 0, E_, 1,                 // B: G^T access  (k stride 1, n stride E)
                 (long)C_ * E_, E_, 0, 1, 0, 0);
            {
                int tot = B_ * E_;
                k_hgout<<<(tot + 255) / 256, 256, 0, stream>>>(ef, xw,
                    hg_w[d] + (size_t)L * 2 * C_, hg_b[d] + L, vmat[d]);
            }
        }

        // ---- build dynamic adjacencies ----
        k_diagfill<<<(B_ * N_ * N_ + 255) / 256, 256, 0, stream>>>(adp, adpF, adpB);
        k_scatter<<<(B_ * E_ + 255) / 256, 256, 0, stream>>>(adpF, adpB, adp,
                                                             idx, idy, vmat[0], vmat[1]);

        // ---- tanh*sigmoid dilated conv -> zcat group 0 ----
        {
            int tot = B_ * C_ * N_ * Tp;
            k_filtgate<<<(tot + 255) / 256, 256, 0, stream>>>(hcur,
                filtw + (size_t)L * C_ * C_ * 2, filtb + (size_t)L * C_,
                gatew + (size_t)L * C_ * C_ * 2, gateb + (size_t)L * C_,
                zcat, T, Tp);
        }

        // ---- 6 graph-diffusion GEMMs (order 2, three adjacencies) ----
        {
            long zB = (long)7 * C_ * N_ * Tp;
            const float* As[6] = {adp, adp, adpF, adpF, adpB, adpB};
            long aBs[6] = {0, 0, (long)N_ * N_, (long)N_ * N_, (long)N_ * N_, (long)N_ * N_};
            int src[6] = {0, 1, 0, 3, 0, 5};
            int dst[6] = {1, 2, 3, 4, 5, 6};
            for (int s = 0; s < 6; ++s) {
                gemm(stream, As[s],
                     zcat + (size_t)src[s] * C_ * N_ * Tp,
                     zcat + (size_t)dst[s] * C_ * N_ * Tp,
                     nullptr, nullptr,
                     N_, C_ * Tp, N_, B_,
                     aBs[s], zB, zB,
                     N_, 0, N_, 1,                    // A[v,w] row-major (vector path)
                     Tp, (long)N_ * Tp, 1, Tp,        // B: z[b,c,w,t], k=w, n=(c,t)
                     0, Tp, (long)N_ * Tp, 1,         // C: same layout, m=v
                     0, 0);
            }
        }

        // ---- 7C dilated conv -> zdg ----
        {
            int tot = B_ * C_ * N_ * Tpp;
            k_dg<<<(tot + 255) / 256, 256, 0, stream>>>(zcat,
                dgw + (size_t)L * C_ * 7 * C_ * 2, dgb + (size_t)L * C_, zdg, Tp, Tpp);
        }

        // ---- skip contribution: only the last time column survives to output ----
        gemm(stream, skipw + (size_t)L * SK_ * C_, zdg + (Tpp - 1), skipf,
             skipb + (size_t)L * SK_, nullptr,
             SK_, N_, C_, B_,
             0, (long)C_ * N_ * Tpp, (long)SK_ * N_,
             SK_, 0, C_, 1,
             N_, 0, Tpp, (long)N_ * Tpp,      // B: zdg[b,c,n,Tpp-1], k=c, n stride Tpp
             0, N_, 0, 1,
             0, 1);                            // accumulate into skipf
        // ---- residual -> next h ----
        {
            int tot = B_ * C_ * N_ * Tpp;
            k_residual<<<(tot + 255) / 256, 256, 0, stream>>>(zdg, hcur, hnext, T, Tpp);
        }
        float* tmp = hcur; hcur = hnext; hnext = tmp;
    }

    // ---- output head ----
    k_relu<<<(B_ * SK_ * N_ + 255) / 256, 256, 0, stream>>>(skipf, B_ * SK_ * N_);
    gemm(stream, end1w, skipf, e1, end1b, nullptr,
         EC_, N_, SK_, B_,
         0, (long)SK_ * N_, (long)EC_ * N_,
         EC_, 0, SK_, 1,
         N_, 0, 1, N_,
         0, N_, 0, 1,
         1, 0);                                // relu
    gemm(stream, end2w, e1, out, end2b, nullptr,
         OD_, N_, EC_, B_,
         0, (long)EC_ * N_, (long)OD_ * N_,
         OD_, 0, EC_, 1,
         N_, 0, 1, N_,
         0, N_, 0, 1,
         0, 0);
}
